// HST_Block_82068235092122
// MI455X (gfx1250) — matrix operand order repaired and checked by reading, not compiled, not run on hardware
//
#include <hip/hip_runtime.h>
#include <math.h>

// ---------------------------------------------------------------------------
// Swin transformer pair (window attn + MLP, shifted window attn + MLP)
// B=16, H=W=56, C=384, HEADS=12, d=32, WIN=7, N=49 (padded to 64), SHIFT=3
// All matmuls: f16 inputs, f32 accumulate via v_wmma_f32_16x16x32_f16.
// GEMMs are double-buffered through LDS with register staging so global
// loads for tile k+1 overlap the WMMA chain for tile k.  All fragments for a
// K-step are loaded into distinct registers before the WMMA chain so LDS
// latency overlaps matrix math instead of gating each WMMA.
// ---------------------------------------------------------------------------

typedef __attribute__((ext_vector_type(16))) _Float16 v16h;
typedef __attribute__((ext_vector_type(8)))  float    v8f;

union AF { v16h v; unsigned int u[8]; _Float16 h[16]; };
union BF { v16h v; uint4 q[2];        _Float16 h[16]; };

__device__ inline v8f wmma_f16(v16h a, v16h b, v8f c) {
  // D = A(16x32 f16) * B(32x16 f16) + C(16x16 f32)
  return __builtin_amdgcn_wmma_f32_16x16x32_f16(
      /*neg_a=*/false, a, /*neg_b=*/false, b,
      /*c_mod=*/(short)0, c, /*reuse_a=*/false, /*reuse_b=*/false);
}

// ---------------------------------------------------------------------------
// Weight convert + transpose: W[K][N] fp32 -> WT[N][K] f16
// ---------------------------------------------------------------------------
__global__ void convT_kernel(const float* __restrict__ W, _Float16* __restrict__ WT,
                             int K, int N) {
  int idx = blockIdx.x * 256 + threadIdx.x;
  if (idx >= K * N) return;
  int k = idx / N, n = idx % N;
  WT[(size_t)n * K + k] = (_Float16)W[idx];
}

// ---------------------------------------------------------------------------
// LayerNorm (+ optional window-partition with cyclic shift, zero-padded to 64
// rows/window). One wave per row; C=384 -> 12 elements/lane.
// ---------------------------------------------------------------------------
__global__ __launch_bounds__(128) void ln_kernel(
    const float* __restrict__ src, const float* __restrict__ g,
    const float* __restrict__ b, _Float16* __restrict__ dst,
    int rows, int windowed, int shift) {
  int row  = blockIdx.x * 4 + (threadIdx.x >> 5);
  int lane = threadIdx.x & 31;
  if (row >= rows) return;

  long t;
  if (windowed) {
    int r = row & 63, win = row >> 6;
    if (r >= 49) {  // zero padding rows
      for (int i = 0; i < 12; ++i)
        dst[(size_t)row * 384 + lane + i * 32] = (_Float16)0.0f;
      return;
    }
    int widx = win & 63, bb = win >> 6;
    int wh = widx >> 3, ww = widx & 7;
    int ii = r / 7, jj = r % 7;
    int hh = (wh * 7 + ii + shift) % 56;
    int wc = (ww * 7 + jj + shift) % 56;
    t = (long)bb * 3136 + hh * 56 + wc;
  } else {
    t = row;
  }

  const float* p = src + (size_t)t * 384;
  float xv[12];
  float s = 0.f;
  #pragma unroll
  for (int i = 0; i < 12; ++i) { xv[i] = p[lane + i * 32]; s += xv[i]; }
  for (int o = 16; o > 0; o >>= 1) s += __shfl_xor(s, o);
  float mean = s * (1.0f / 384.0f);
  float vs = 0.f;
  #pragma unroll
  for (int i = 0; i < 12; ++i) { float d = xv[i] - mean; vs += d * d; }
  for (int o = 16; o > 0; o >>= 1) vs += __shfl_xor(vs, o);
  float rstd = rsqrtf(vs * (1.0f / 384.0f) + 1e-5f);
  #pragma unroll
  for (int i = 0; i < 12; ++i) {
    int c = lane + i * 32;
    float o = (xv[i] - mean) * rstd * g[c] + b[c];
    dst[(size_t)row * 384 + c] = (_Float16)o;
  }
}

// ---------------------------------------------------------------------------
// Generic WMMA GEMM: Out[M,N] = A16[M,K] @ (BT16[N,K])^T + bias, epilogues:
//   mode 0: f16 out, cols<384 scaled by qscale (QKV)
//   mode 1: exact GELU, f16 out (MLP fc1)
//   mode 2: window-reverse(+shift) + fp32 residual add (attn proj); M=padded
//   mode 3: fp32 residual add, token rows (MLP fc2)
// 256 thr (8 waves), 128x128 tile, BK=32, double-buffered LDS + reg staging.
// ---------------------------------------------------------------------------
__global__ __launch_bounds__(256) void gemm_kernel(
    const _Float16* __restrict__ A, const _Float16* __restrict__ BT,
    const float* __restrict__ bias, const float* __restrict__ resid,
    float* __restrict__ outF, _Float16* __restrict__ outH,
    int M, int N, int K, int mode, int shift, float qscale) {
  __shared__ _Float16 lA[2][128 * 48];   // stride 48 halves = 96B (16B align)
  __shared__ _Float16 lB[2][128 * 48];   // lB[n][k] (weights pre-transposed)

  const int tid = threadIdx.x;
  const int m0 = blockIdx.y * 128, n0 = blockIdx.x * 128;
  const int wv = tid >> 5, lane = tid & 31;
  const int wm = wv & 3, wn = wv >> 2;          // 4 x 2 wave grid
  const int hi = lane >> 4, ln15 = lane & 15;

  // staging slot coordinates (2 x uint4 per thread per matrix)
  const int id0 = tid * 2;
  const int row0 = id0 >> 2, c40 = (id0 & 3) * 8;
  const int row1 = (id0 + 1) >> 2, c41 = ((id0 + 1) & 3) * 8;

  v8f zero = {0.f, 0.f, 0.f, 0.f, 0.f, 0.f, 0.f, 0.f};
  v8f acc[2][4];
  #pragma unroll
  for (int i = 0; i < 2; ++i)
    #pragma unroll
    for (int j = 0; j < 4; ++j) acc[i][j] = zero;

  uint4 ra0, ra1, rb0, rb1;
  // prologue: fetch + stage tile 0
  ra0 = *(const uint4*)(A  + (size_t)(m0 + row0) * K + c40);
  ra1 = *(const uint4*)(A  + (size_t)(m0 + row1) * K + c41);
  rb0 = *(const uint4*)(BT + (size_t)(n0 + row0) * K + c40);
  rb1 = *(const uint4*)(BT + (size_t)(n0 + row1) * K + c41);
  *(uint4*)(lA[0] + row0 * 48 + c40) = ra0;
  *(uint4*)(lA[0] + row1 * 48 + c41) = ra1;
  *(uint4*)(lB[0] + row0 * 48 + c40) = rb0;
  *(uint4*)(lB[0] + row1 * 48 + c41) = rb1;
  __syncthreads();

  const int nk = K >> 5;
  for (int kk = 0; kk < nk; ++kk) {
    const int cur = kk & 1, nxt = cur ^ 1;
    if (kk + 1 < nk) {   // issue next tile's global loads early
      int k0 = (kk + 1) * 32;
      ra0 = *(const uint4*)(A  + (size_t)(m0 + row0) * K + k0 + c40);
      ra1 = *(const uint4*)(A  + (size_t)(m0 + row1) * K + k0 + c41);
      rb0 = *(const uint4*)(BT + (size_t)(n0 + row0) * K + k0 + c40);
      rb1 = *(const uint4*)(BT + (size_t)(n0 + row1) * K + k0 + c41);
    }

    // load ALL fragments for this K-step into distinct registers first so
    // the ds_load_b128s issue back-to-back and overlap the WMMA chain
    AF af[2];
    BF bf[4];
    #pragma unroll
    for (int mt = 0; mt < 2; ++mt) {
      int mrow = wm * 32 + mt * 16 + ln15;
      int kb = hi ? 8 : 0;
      #pragma unroll
      for (int i = 0; i < 8; ++i) {       // ISA A-frag layout, K pairs
        int k = ((i < 4) ? kb : (kb + 16)) + (i & 3) * 2;
        af[mt].u[i] = *(const unsigned int*)(lA[cur] + mrow * 48 + k);
      }
    }
    #pragma unroll
    for (int nt = 0; nt < 4; ++nt) {       // ISA B-frag: lane=N col, K as A-frag (8h, 16+8h)
      int ncol = wn * 64 + nt * 16 + ln15;
      int kb = hi ? 8 : 0;
      bf[nt].q[0] = *(const uint4*)(lB[cur] + ncol * 48 + kb);
      bf[nt].q[1] = *(const uint4*)(lB[cur] + ncol * 48 + kb + 16);
    }
    #pragma unroll
    for (int nt = 0; nt < 4; ++nt)
      #pragma unroll
      for (int mt = 0; mt < 2; ++mt)
        acc[mt][nt] = wmma_f16(af[mt].v, bf[nt].v, acc[mt][nt]);

    if (kk + 1 < nk) {   // stage next tile (waits loads here, post-compute)
      *(uint4*)(lA[nxt] + row0 * 48 + c40) = ra0;
      *(uint4*)(lA[nxt] + row1 * 48 + c41) = ra1;
      *(uint4*)(lB[nxt] + row0 * 48 + c40) = rb0;
      *(uint4*)(lB[nxt] + row1 * 48 + c41) = rb1;
      __syncthreads();
    }
  }

  // epilogue: D layout -> row = r + 8*hi, col = ln15
  #pragma unroll
  for (int mt = 0; mt < 2; ++mt) {
    #pragma unroll
    for (int nt = 0; nt < 4; ++nt) {
      int gcol = n0 + wn * 64 + nt * 16 + ln15;
      float bv = bias[gcol];
      #pragma unroll
      for (int r = 0; r < 8; ++r) {
        int grow = m0 + wm * 32 + mt * 16 + r + 8 * hi;
        float v = acc[mt][nt][r] + bv;
        if (mode == 0) {
          if (gcol < 384) v *= qscale;             // scale q by d^-0.5
          outH[(size_t)grow * N + gcol] = (_Float16)v;
        } else if (mode == 1) {
          float gl = 0.5f * v * (1.0f + erff(v * 0.70710678118654752f));
          outH[(size_t)grow * N + gcol] = (_Float16)gl;
        } else if (mode == 2) {
          int rr = grow & 63;
          if (rr < 49) {                           // drop padded rows
            int win = grow >> 6;
            int widx = win & 63, bb = win >> 6;
            int wh = widx >> 3, ww = widx & 7;
            int ii = rr / 7, jj = rr % 7;
            int hh = (wh * 7 + ii + shift) % 56;
            int wc = (ww * 7 + jj + shift) % 56;
            size_t t = (size_t)bb * 3136 + (size_t)hh * 56 + wc;
            outF[t * 384 + gcol] = resid[t * 384 + gcol] + v;
          }
        } else {
          size_t o = (size_t)grow * N + gcol;
          outF[o] = resid[o] + v;
        }
      }
    }
  }
}

// ---------------------------------------------------------------------------
// Windowed attention: one workgroup (4 waves) per (window, head).
// S = q k^T (4 WMMA/wave) + rel-pos bias (+ shift mask), softmax over valid
// keys (<49), P f16 -> LDS, O = P V (4 WMMA/wave).  V is transposed into LDS
// at load time so O-phase B-frags are 2x ds_load_b128.  Bias column for this
// head is staged in LDS.
// ---------------------------------------------------------------------------
__global__ __launch_bounds__(128) void attn_kernel(
    const _Float16* __restrict__ qkv,    // [win*64][1152]: q|k|v, q pre-scaled
    const float* __restrict__ biasTab,   // [169][12]
    _Float16* __restrict__ out,          // [win*64][384]
    int useMask) {
  __shared__ _Float16 qs[64 * 48];
  __shared__ _Float16 ks[64 * 48];
  __shared__ _Float16 vsT[32 * 72];      // vsT[d][key], stride 72 (16B align)
  __shared__ _Float16 ps[64 * 66];
  __shared__ float    bT[176];           // bias[relidx] for this head

  const int win = blockIdx.x, head = blockIdx.y;
  const int tid = threadIdx.x;

  for (int i = tid; i < 169; i += 128) bT[i] = biasTab[i * 12 + head];

  #pragma unroll
  for (int j = 0; j < 2; ++j) {       // 128 thr * 2 uint4 = 64x32 halves each
    int id = tid * 2 + j;
    int row = id >> 2, c4 = (id & 3) * 8;
    size_t base = ((size_t)win * 64 + row) * 1152 + head * 32 + c4;
    *(uint4*)(qs + row * 48 + c4) = *(const uint4*)(qkv + base);
    *(uint4*)(ks + row * 48 + c4) = *(const uint4*)(qkv + base + 384);
    uint4 vq = *(const uint4*)(qkv + base + 768);
    const _Float16* hp = (const _Float16*)&vq;
    #pragma unroll
    for (int i = 0; i < 8; ++i)       // transpose V: [key][d] -> [d][key]
      vsT[(c4 + i) * 72 + row] = hp[i];
  }
  __syncthreads();

  const int wv = tid >> 5, lane = tid & 31;
  const int hi = lane >> 4, ln15 = lane & 15;
  const int widx = win & 63;
  const int wh = widx >> 3, ww = widx & 7;

  // ---- S = q @ k^T : wave wv owns query rows [16wv, 16wv+16) ----
  v8f zero = {0.f, 0.f, 0.f, 0.f, 0.f, 0.f, 0.f, 0.f};
  v8f accS[4];
  #pragma unroll
  for (int nt = 0; nt < 4; ++nt) accS[nt] = zero;

  AF af;
  BF bfS[4];
  {
    int mrow = 16 * wv + ln15;
    int kb = hi ? 8 : 0;
    #pragma unroll
    for (int i = 0; i < 8; ++i) {
      int k = ((i < 4) ? kb : (kb + 16)) + (i & 3) * 2;
      af.u[i] = *(const unsigned int*)(qs + mrow * 48 + k);
    }
  }
  #pragma unroll
  for (int nt = 0; nt < 4; ++nt) {
    int ncol = nt * 16 + ln15;           // key index
    int kb = hi ? 8 : 0;                 // over head-dim
    bfS[nt].q[0] = *(const uint4*)(ks + ncol * 48 + kb);
    bfS[nt].q[1] = *(const uint4*)(ks + ncol * 48 + kb + 16);
  }
  #pragma unroll
  for (int nt = 0; nt < 4; ++nt)
    accS[nt] = wmma_f16(af.v, bfS[nt].v, accS[nt]);

  // ---- bias + mask + softmax (per row, across 16-lane half-groups) ----
  #pragma unroll
  for (int r = 0; r < 8; ++r) {
    int prow = 16 * wv + r + 8 * hi;     // query row in window (0..63)
    int qi = prow / 7, qj = prow % 7;
    float vv[4];
    #pragma unroll
    for (int nt = 0; nt < 4; ++nt) {
      int kcol = nt * 16 + ln15;
      float xv = accS[nt][r];
      if (kcol >= 49) {
        xv = -1e30f;                     // padded key -> excluded
      } else if (prow < 49) {
        int ki = kcol / 7, kj = kcol % 7;
        int ridx = (qi - ki + 6) * 13 + (qj - kj + 6);
        xv += bT[ridx];
        if (useMask) {                   // shift-window region mask
          int hq = wh * 7 + qi, wq = ww * 7 + qj;
          int hk = wh * 7 + ki, wk = ww * 7 + kj;
          int rq = 3 * ((hq < 49) ? 0 : ((hq < 53) ? 1 : 2)) +
                       ((wq < 49) ? 0 : ((wq < 53) ? 1 : 2));
          int rk = 3 * ((hk < 49) ? 0 : ((hk < 53) ? 1 : 2)) +
                       ((wk < 49) ? 0 : ((wk < 53) ? 1 : 2));
          if (rq != rk) xv += -100.0f;
        }
      }
      vv[nt] = xv;
    }
    float mx = fmaxf(fmaxf(vv[0], vv[1]), fmaxf(vv[2], vv[3]));
    for (int o = 8; o > 0; o >>= 1) mx = fmaxf(mx, __shfl_xor(mx, o));
    float s = 0.f;
    #pragma unroll
    for (int nt = 0; nt < 4; ++nt) { vv[nt] = __expf(vv[nt] - mx); s += vv[nt]; }
    for (int o = 8; o > 0; o >>= 1) s += __shfl_xor(s, o);
    float inv = 1.0f / s;
    #pragma unroll
    for (int nt = 0; nt < 4; ++nt)
      ps[prow * 66 + nt * 16 + ln15] = (_Float16)(vv[nt] * inv);
  }
  __syncthreads();

  // ---- O = P @ V : wave wv -> rows [16wv,16wv+16) x d(0..31) ----
  v8f accO[2];
  accO[0] = zero; accO[1] = zero;
  #pragma unroll
  for (int kc = 0; kc < 2; ++kc) {       // K=64 keys, 2 chunks of 32
    AF pa;
    BF vb[2];
    int mrow = 16 * wv + ln15;
    int kb = hi ? 8 : 0;
    #pragma unroll
    for (int i = 0; i < 8; ++i) {
      int k = kc * 32 + ((i < 4) ? kb : (kb + 16)) + (i & 3) * 2;
      pa.u[i] = *(const unsigned int*)(ps + mrow * 66 + k);
    }
    #pragma unroll
    for (int ntile = 0; ntile < 2; ++ntile) {  // from transposed V: K as A-frag (8h, 16+8h)
      int d = ntile * 16 + ln15;
      int kb2 = kc * 32 + (hi ? 8 : 0);
      vb[ntile].q[0] = *(const uint4*)(vsT + d * 72 + kb2);
      vb[ntile].q[1] = *(const uint4*)(vsT + d * 72 + kb2 + 16);
    }
    #pragma unroll
    for (int ntile = 0; ntile < 2; ++ntile)
      accO[ntile] = wmma_f16(pa.v, vb[ntile].v, accO[ntile]);
  }
  #pragma unroll
  for (int ntile = 0; ntile < 2; ++ntile) {
    #pragma unroll
    for (int r = 0; r < 8; ++r) {
      int prow = 16 * wv + r + 8 * hi;
      int col = head * 32 + ntile * 16 + ln15;
      out[((size_t)win * 64 + prow) * 384 + col] = (_Float16)accO[ntile][r];
    }
  }
}

// ---------------------------------------------------------------------------
// Host orchestration
// ---------------------------------------------------------------------------
extern "C" void kernel_launch(void* const* d_in, const int* in_sizes, int n_in,
                              void* d_out, int out_size, void* d_ws, size_t ws_size,
                              hipStream_t stream) {
  (void)in_sizes; (void)n_in; (void)out_size; (void)ws_size;

  const float* x       = (const float*)d_in[0];
  const float* ln1_g   = (const float*)d_in[1];
  const float* ln1_b   = (const float*)d_in[2];
  const float* qkv1_w  = (const float*)d_in[3];
  const float* qkv1_b  = (const float*)d_in[4];
  const float* proj1_w = (const float*)d_in[5];
  const float* proj1_b = (const float*)d_in[6];
  const float* bias1   = (const float*)d_in[7];
  const float* ln2_g   = (const float*)d_in[8];
  const float* ln2_b   = (const float*)d_in[9];
  const float* m1w1    = (const float*)d_in[10];
  const float* m1b1    = (const float*)d_in[11];
  const float* m1w2    = (const float*)d_in[12];
  const float* m1b2    = (const float*)d_in[13];
  const float* ln3_g   = (const float*)d_in[14];
  const float* ln3_b   = (const float*)d_in[15];
  const float* qkv2_w  = (const float*)d_in[16];
  const float* qkv2_b  = (const float*)d_in[17];
  const float* proj2_w = (const float*)d_in[18];
  const float* proj2_b = (const float*)d_in[19];
  const float* bias2   = (const float*)d_in[20];
  const float* ln4_g   = (const float*)d_in[21];
  const float* ln4_b   = (const float*)d_in[22];
  const float* m2w1    = (const float*)d_in[23];
  const float* m2b1    = (const float*)d_in[24];
  const float* m2w2    = (const float*)d_in[25];
  const float* m2b2    = (const float*)d_in[26];

  // workspace layout (bytes)
  char* ws = (char*)d_ws;
  const size_t SZ_QKVW = 1152u * 384u * 2u;     //  884736
  const size_t SZ_PRJW = 384u * 384u * 2u;      //  294912
  const size_t SZ_W1   = 1536u * 384u * 2u;     // 1179648
  const size_t WBLK    = SZ_QKVW + SZ_PRJW + 2 * SZ_W1;  // 3538944
  _Float16* qkv1T = (_Float16*)(ws);
  _Float16* prj1T = (_Float16*)(ws + SZ_QKVW);
  _Float16* w11T  = (_Float16*)(ws + SZ_QKVW + SZ_PRJW);
  _Float16* w12T  = (_Float16*)(ws + SZ_QKVW + SZ_PRJW + SZ_W1);
  _Float16* qkv2T = (_Float16*)(ws + WBLK);
  _Float16* prj2T = (_Float16*)(ws + WBLK + SZ_QKVW);
  _Float16* w21T  = (_Float16*)(ws + WBLK + SZ_QKVW + SZ_PRJW);
  _Float16* w22T  = (_Float16*)(ws + WBLK + SZ_QKVW + SZ_PRJW + SZ_W1);

  const size_t OFF_XW  = 2 * WBLK;                       //   7077888
  const size_t OFF_U   = OFF_XW + 65536ull * 384 * 2;    //  57409536
  const size_t OFF_ATT = OFF_U + 50176ull * 1536 * 2;    // 211550208
  const size_t OFF_X1  = OFF_ATT + 65536ull * 384 * 2;   // 261881856
  const size_t OFF_X2  = OFF_X1 + 50176ull * 384 * 4;    // 338952192
  _Float16* xw16  = (_Float16*)(ws + OFF_XW);   // LN out (padded win / token)
  _Float16* ubuf  = (_Float16*)(ws + OFF_U);    // qkv16 OR mlp-hidden f16
  _Float16* att16 = (_Float16*)(ws + OFF_ATT);  // attention output f16
  float*    x1    = (float*)(ws + OFF_X1);      // residual stream (also x3)
  float*    x2    = (float*)(ws + OFF_X2);
  float*    outF  = (float*)d_out;

  const float QS = 0.17677669529663687f;  // 1/sqrt(32)
  const int TP = 65536, T = 50176;

  // --- convert + transpose all weights to f16 [N][K] ---
  convT_kernel<<<dim3((384 * 1152 + 255) / 256), 256, 0, stream>>>(qkv1_w, qkv1T, 384, 1152);
  convT_kernel<<<dim3((384 * 384 + 255) / 256), 256, 0, stream>>>(proj1_w, prj1T, 384, 384);
  convT_kernel<<<dim3((384 * 1536 + 255) / 256), 256, 0, stream>>>(m1w1, w11T, 384, 1536);
  convT_kernel<<<dim3((1536 * 384 + 255) / 256), 256, 0, stream>>>(m1w2, w12T, 1536, 384);
  convT_kernel<<<dim3((384 * 1152 + 255) / 256), 256, 0, stream>>>(qkv2_w, qkv2T, 384, 1152);
  convT_kernel<<<dim3((384 * 384 + 255) / 256), 256, 0, stream>>>(proj2_w, prj2T, 384, 384);
  convT_kernel<<<dim3((384 * 1536 + 255) / 256), 256, 0, stream>>>(m2w1, w21T, 384, 1536);
  convT_kernel<<<dim3((1536 * 384 + 255) / 256), 256, 0, stream>>>(m2w2, w22T, 1536, 384);

  // ================= Block 1 (no shift, no mask) =================
  ln_kernel<<<dim3(TP / 4), 128, 0, stream>>>(x, ln1_g, ln1_b, xw16, TP, 1, 0);
  gemm_kernel<<<dim3(1152 / 128, TP / 128), 256, 0, stream>>>(
      xw16, qkv1T, qkv1_b, nullptr, nullptr, ubuf, TP, 1152, 384, 0, 0, QS);
  attn_kernel<<<dim3(1024, 12), 128, 0, stream>>>(ubuf, bias1, att16, 0);
  gemm_kernel<<<dim3(384 / 128, TP / 128), 256, 0, stream>>>(
      att16, prj1T, proj1_b, x, x1, nullptr, TP, 384, 384, 2, 0, 1.f);
  ln_kernel<<<dim3(T / 4), 128, 0, stream>>>(x1, ln2_g, ln2_b, xw16, T, 0, 0);
  gemm_kernel<<<dim3(1536 / 128, T / 128), 256, 0, stream>>>(
      xw16, w11T, m1b1, nullptr, nullptr, ubuf, T, 1536, 384, 1, 0, 1.f);
  gemm_kernel<<<dim3(384 / 128, T / 128), 256, 0, stream>>>(
      ubuf, w12T, m1b2, x1, x2, nullptr, T, 384, 1536, 3, 0, 1.f);

  // ================= Block 2 (shift=3, masked) =================
  ln_kernel<<<dim3(TP / 4), 128, 0, stream>>>(x2, ln3_g, ln3_b, xw16, TP, 1, 3);
  gemm_kernel<<<dim3(1152 / 128, TP / 128), 256, 0, stream>>>(
      xw16, qkv2T, qkv2_b, nullptr, nullptr, ubuf, TP, 1152, 384, 0, 0, QS);
  attn_kernel<<<dim3(1024, 12), 128, 0, stream>>>(ubuf, bias2, att16, 1);
  gemm_kernel<<<dim3(384 / 128, TP / 128), 256, 0, stream>>>(
      att16, prj2T, proj2_b, x2, x1 /*x3*/, nullptr, TP, 384, 384, 2, 3, 1.f);
  ln_kernel<<<dim3(T / 4), 128, 0, stream>>>(x1, ln4_g, ln4_b, xw16, T, 0, 0);
  gemm_kernel<<<dim3(1536 / 128, T / 128), 256, 0, stream>>>(
      xw16, w21T, m2b1, nullptr, nullptr, ubuf, T, 1536, 384, 1, 0, 1.f);
  gemm_kernel<<<dim3(384 / 128, T / 128), 256, 0, stream>>>(
      ubuf, w22T, m2b2, x1, outF, nullptr, T, 384, 1536, 3, 0, 1.f);
}
